// Transformer_8169027797217
// MI455X (gfx1250) — matrix operand order repaired and checked by reading, not compiled, not run on hardware
//
#include <hip/hip_runtime.h>
#include <hip/hip_bf16.h>
#include <math.h>

// ---------------------------------------------------------------------------
// Types for CDNA5 WMMA (wave32): 16x16x32 f16 -> f32
// ---------------------------------------------------------------------------
typedef _Float16 v16h __attribute__((ext_vector_type(16)));
typedef float    v8f  __attribute__((ext_vector_type(8)));
typedef _Float16 h4   __attribute__((ext_vector_type(4)));

union FragH {
    v16h     v;
    uint4    q[2];
    _Float16 h[16];
};

__device__ __forceinline__ v8f wmma_16x16x32(v16h a, v16h b, v8f c) {
    // D = A(16x32 f16) x B(32x16 f16) + C(16x16 f32)
    return __builtin_amdgcn_wmma_f32_16x16x32_f16(
        /*neg_a=*/false, a, /*neg_b=*/false, b,
        /*c_mod=*/(short)0, c, /*reuse_a=*/false, /*reuse_b=*/false);
}

__device__ __forceinline__ v8f zero_v8f() {
    v8f z;
#pragma unroll
    for (int i = 0; i < 8; ++i) z[i] = 0.0f;
    return z;
}

__device__ __forceinline__ h4 cvt_h4(float4 f) {
    h4 r;
    r[0] = (_Float16)f.x; r[1] = (_Float16)f.y;
    r[2] = (_Float16)f.z; r[3] = (_Float16)f.w;
    return r;
}

// ---------------------------------------------------------------------------
// GEMM: C[M,N] = A[M,K] * W[K,N] + bias[N]  (optional ReLU)
//
// Weights may be stored as consecutive [K x bChunkCols] chunks of pitch
// bPitch elements (handles the [4,D,D] packed QKV weight layout). A is f32
// row-major (lda); C is dense f32 row-major (row stride N). f32 -> f16
// conversion fused into LDS staging.
//
// Macro tile 128x128, BK=32, 256 threads = 8 waves in a 2(M) x 4(N) grid;
// each wave computes 64x32 = 4x2 WMMA tiles. Software-pipelined, LDS
// double-buffered: next tile's global loads (b128) are in flight while the
// current tile's WMMAs execute.
// ---------------------------------------------------------------------------
#define BM 128
#define BN 128
#define BK 32
#define LDT 48   // LDS row stride in halfs (96B: keeps b128 frag reads 16B-aligned)

__global__ __launch_bounds__(256) void k_gemm(
    const float* __restrict__ A, int lda,
    const float* __restrict__ W, long long bPitch, int bChunkCols,
    const float* __restrict__ bias,
    float* __restrict__ C, int N, int K, int relu)
{
    __shared__ __align__(16) _Float16 As[2][BM * LDT];
    __shared__ __align__(16) _Float16 Bs[2][BN * LDT];

    const int tid = threadIdx.x;
    const int n0  = blockIdx.x * BN;
    const int m0  = blockIdx.y * BM;

    // chunked weight addressing (bChunkCols is a multiple of BN, so the
    // whole 128-wide tile lives in a single chunk)
    const int chunk = n0 / bChunkCols;
    const float* Wbase = W + (size_t)chunk * (size_t)bPitch + (n0 % bChunkCols);

    const int w    = tid >> 5;
    const int lane = tid & 31;
    const int wm   = (w >> 2) * 64;   // wave row offset within tile
    const int wn   = (w & 3) * 32;    // wave col offset within tile
    const int lm   = lane & 15;
    const int lh   = lane >> 4;

    // per-thread staging coordinates (4 float4 each for A and B tiles)
    // A: 128 rows x 32 cols  -> idx: row = idx>>3, col4 = (idx&7)*4
    // B:  32 krows x 128 cols-> idx: kk  = idx>>5, n4   = (idx&31)*4
    int aRow[4], aC4[4], bKk[4], bN4[4];
#pragma unroll
    for (int i = 0; i < 4; ++i) {
        int idx = i * 256 + tid;
        aRow[i] = idx >> 3;  aC4[i] = (idx & 7) * 4;
        bKk[i]  = idx >> 5;  bN4[i] = (idx & 31) * 4;
    }

    v8f acc[4][2];
#pragma unroll
    for (int mt = 0; mt < 4; ++mt)
#pragma unroll
        for (int nt = 0; nt < 2; ++nt) acc[mt][nt] = zero_v8f();

    float4 aReg[4], bReg[4];

    auto loadTile = [&](int k0) {
#pragma unroll
        for (int i = 0; i < 4; ++i)
            aReg[i] = *(const float4*)&A[(size_t)(m0 + aRow[i]) * lda + k0 + aC4[i]];
#pragma unroll
        for (int i = 0; i < 4; ++i)
            bReg[i] = *(const float4*)&Wbase[(size_t)(k0 + bKk[i]) * bChunkCols + bN4[i]];
        // prefetch one more K tile ahead (global_prefetch_b8)
        if (k0 + BK < K) {
            if (tid < BM)
                __builtin_prefetch(&A[(size_t)(m0 + tid) * lda + k0 + BK], 0, 1);
            if (tid >= 128 && tid < 128 + BK)
                __builtin_prefetch(&Wbase[(size_t)(k0 + BK + tid - 128) * bChunkCols], 0, 1);
        }
    };

    auto storeTile = [&](int buf) {
#pragma unroll
        for (int i = 0; i < 4; ++i)
            *(h4*)&As[buf][aRow[i] * LDT + aC4[i]] = cvt_h4(aReg[i]);
#pragma unroll
        for (int i = 0; i < 4; ++i) {
            h4 hv = cvt_h4(bReg[i]);
#pragma unroll
            for (int j = 0; j < 4; ++j)
                Bs[buf][(bN4[i] + j) * LDT + bKk[i]] = hv[j];   // transposed: Bs[n][k]
        }
    };

    auto compute = [&](int buf) {
        FragH af[4], bf[2];
#pragma unroll
        for (int mt = 0; mt < 4; ++mt) {
            int base = (wm + mt * 16 + lm) * LDT + lh * 8;
            af[mt].q[0] = *(const uint4*)&As[buf][base];        // K = lh*8 + 0..7
            af[mt].q[1] = *(const uint4*)&As[buf][base + 16];   // K = 16 + lh*8 + 0..7
        }
#pragma unroll
        for (int nt = 0; nt < 2; ++nt) {
            int base = (wn + nt * 16 + lm) * LDT + lh * 8;
            bf[nt].q[0] = *(const uint4*)&Bs[buf][base];        // K = lh*8 + 0..7
            bf[nt].q[1] = *(const uint4*)&Bs[buf][base + 16];   // K = 16 + lh*8 + 0..7
        }
#pragma unroll
        for (int mt = 0; mt < 4; ++mt)
#pragma unroll
            for (int nt = 0; nt < 2; ++nt)
                acc[mt][nt] = wmma_16x16x32(af[mt].v, bf[nt].v, acc[mt][nt]);
    };

    // ---- software pipeline: prologue ----
    loadTile(0);
    storeTile(0);
    __syncthreads();

    const int nK = K / BK;
    for (int kt = 0; kt < nK; ++kt) {
        if (kt + 1 < nK) loadTile((kt + 1) * BK);   // global loads in flight
        compute(kt & 1);                            // WMMAs hide load latency
        if (kt + 1 < nK) storeTile((kt + 1) & 1);   // cvt + LDS store
        __syncthreads();
    }

    // ---- epilogue: bias (+ReLU), f32 store ----
#pragma unroll
    for (int mt = 0; mt < 4; ++mt) {
#pragma unroll
        for (int nt = 0; nt < 2; ++nt) {
            int gn = n0 + wn + nt * 16 + lm;
            float bv = bias[gn];
#pragma unroll
            for (int r = 0; r < 8; ++r) {
                int gm = m0 + wm + mt * 16 + lh * 8 + r;
                float vv = acc[mt][nt][r] + bv;
                if (relu) vv = fmaxf(vv, 0.0f);
                C[(size_t)gm * N + gn] = vv;
            }
        }
    }
}

// ---------------------------------------------------------------------------
// Flash attention: one wave per (b, h, 16-query tile).
// Streams keys in blocks of 32 with online softmax; S x S scores never
// touch memory. scale = 1/sqrt(512) (faithful to reference).
// ---------------------------------------------------------------------------
__global__ __launch_bounds__(32) void k_attn(
    const float* __restrict__ qB, int qStride,
    const float* __restrict__ kB, const float* __restrict__ vB, int kvStride,
    float* __restrict__ outB, int Skv, float scale)
{
    __shared__ __align__(16) _Float16 Ps[16 * LDT];  // P: C-frag -> A-frag bounce

    const int lane = threadIdx.x;
    const int lm = lane & 15, lh = lane >> 4;
    const int qt = blockIdx.x, hh = blockIdx.y, bb = blockIdx.z;
    const size_t qRow0 = (size_t)bb * 512 + qt * 16;
    const size_t kRow0 = (size_t)bb * 512;
    const int hOff = hh * 64;

    // Q fragments: two 16x32 A-frags covering d=0..31 and d=32..63
    FragH qf[2];
#pragma unroll
    for (int f = 0; f < 2; ++f) {
        const float4* p =
            (const float4*)(qB + (qRow0 + lm) * qStride + hOff + f * 32 + lh * 8);
        float4 a0 = p[0], a1 = p[1], a2 = p[4], a3 = p[5];
        *(h4*)&qf[f].h[0]  = cvt_h4(a0);
        *(h4*)&qf[f].h[4]  = cvt_h4(a1);
        *(h4*)&qf[f].h[8]  = cvt_h4(a2);
        *(h4*)&qf[f].h[12] = cvt_h4(a3);
    }

    v8f o[4];                    // O accumulators: 16 x 64 (four n-tiles of 16)
    float rm[8], rl[8];          // row max / row sum (row m = lh*8 + r)
#pragma unroll
    for (int t = 0; t < 4; ++t) o[t] = zero_v8f();
#pragma unroll
    for (int r = 0; r < 8; ++r) { rm[r] = -3.0e38f; rl[r] = 0.0f; }

    for (int kb = 0; kb < Skv; kb += 32) {
        // ---- S = Q K^T for 32 keys: two 16x16 score frags ----
        v8f s[2]; s[0] = zero_v8f(); s[1] = zero_v8f();
#pragma unroll
        for (int p = 0; p < 2; ++p) {
            FragH kf[2];   // B-frags: B[k=d][n=key] = K[key][d]
#pragma unroll
            for (int f = 0; f < 2; ++f) {
                const float4* kp = (const float4*)(kB + (kRow0 + kb + p * 16 + lm) * kvStride
                                                   + hOff + f * 32 + lh * 8);
                float4 c0 = kp[0], c1 = kp[1], c2 = kp[4], c3 = kp[5];
                *(h4*)&kf[f].h[0]  = cvt_h4(c0);
                *(h4*)&kf[f].h[4]  = cvt_h4(c1);
                *(h4*)&kf[f].h[8]  = cvt_h4(c2);
                *(h4*)&kf[f].h[12] = cvt_h4(c3);
            }
            s[p] = wmma_16x16x32(qf[0].v, kf[0].v, s[p]);
            s[p] = wmma_16x16x32(qf[1].v, kf[1].v, s[p]);
        }

        // ---- online softmax (row reductions across the 16-lane half) ----
        float corr[8];
#pragma unroll
        for (int r = 0; r < 8; ++r) {
            float a0 = s[0][r] * scale;
            float a1 = s[1][r] * scale;
            float mx = fmaxf(a0, a1);
#pragma unroll
            for (int off = 1; off < 16; off <<= 1)
                mx = fmaxf(mx, __shfl_xor(mx, off));
            float nm = fmaxf(rm[r], mx);
            corr[r] = __expf(rm[r] - nm);
            float p0 = __expf(a0 - nm);
            float p1 = __expf(a1 - nm);
            s[0][r] = p0; s[1][r] = p1;
            float sum = p0 + p1;
#pragma unroll
            for (int off = 1; off < 16; off <<= 1)
                sum += __shfl_xor(sum, off);
            rl[r] = rl[r] * corr[r] + sum;
            rm[r] = nm;
        }
#pragma unroll
        for (int t = 0; t < 4; ++t)
#pragma unroll
            for (int r = 0; r < 8; ++r) o[t][r] *= corr[r];

        // ---- re-layout P (C-frag) -> A-frag via LDS ----
        __syncthreads();
#pragma unroll
        for (int r = 0; r < 8; ++r) {
            Ps[(lh * 8 + r) * LDT + lm]      = (_Float16)s[0][r];
            Ps[(lh * 8 + r) * LDT + 16 + lm] = (_Float16)s[1][r];
        }
        __syncthreads();
        FragH pf;
        {
            int base = lm * LDT + lh * 8;
            pf.q[0] = *(const uint4*)&Ps[base];
            pf.q[1] = *(const uint4*)&Ps[base + 16];
        }

        // ---- build all four V B-frags in one batched pass over key rows ----
        FragH vf[4];   // B[k=key][n=d] = V[key][d], n-tiles d = t*16
#pragma unroll
        for (int j = 0; j < 16; ++j) {
            const float* vr = vB + (kRow0 + kb + (j < 8 ? lh * 8 + j : 16 + lh * 8 + (j - 8))) * kvStride + hOff;
#pragma unroll
            for (int t = 0; t < 4; ++t)
                vf[t].h[j] = (_Float16)vr[t * 16 + lm];
        }
#pragma unroll
        for (int t = 0; t < 4; ++t)
            o[t] = wmma_16x16x32(pf.v, vf[t].v, o[t]);
    }

    // ---- normalize and store ----
    float rinv[8];
#pragma unroll
    for (int r = 0; r < 8; ++r) rinv[r] = 1.0f / rl[r];
#pragma unroll
    for (int t = 0; t < 4; ++t)
#pragma unroll
        for (int r = 0; r < 8; ++r)
            outB[(qRow0 + lh * 8 + r) * 512 + hOff + t * 16 + lm] =
                o[t][r] * rinv[r];
}

// ---------------------------------------------------------------------------
// out = LN(a (+ r)) * alpha + beta, per row of D=512.
// Reference semantics: unbiased std (ddof=1), NO epsilon.
// One wave per row; 8 rows per 256-thread block; float4 throughout.
// ---------------------------------------------------------------------------
__global__ __launch_bounds__(256) void k_add_ln(
    const float* __restrict__ a, const float* __restrict__ r,
    const float* __restrict__ alpha, const float* __restrict__ beta,
    float* __restrict__ out, int hasR)
{
    const int w = threadIdx.x >> 5;
    const int lane = threadIdx.x & 31;
    const size_t row = (size_t)blockIdx.x * 8 + w;
    const float4* ap = (const float4*)(a + row * 512);
    const float4* rp = (const float4*)(r + row * 512);

    float4 x[4];
    float s = 0.0f;
#pragma unroll
    for (int q = 0; q < 4; ++q) {
        float4 v = ap[q * 32 + lane];           // lane-contiguous: coalesced
        if (hasR) {
            float4 u = rp[q * 32 + lane];
            v.x += u.x; v.y += u.y; v.z += u.z; v.w += u.w;
        }
        x[q] = v;
        s += v.x + v.y + v.z + v.w;
    }
#pragma unroll
    for (int off = 1; off < 32; off <<= 1) s += __shfl_xor(s, off);
    const float mean = s * (1.0f / 512.0f);

    float vs = 0.0f;
#pragma unroll
    for (int q = 0; q < 4; ++q) {
        float dx = x[q].x - mean, dy = x[q].y - mean,
              dz = x[q].z - mean, dw = x[q].w - mean;
        vs += dx * dx + dy * dy + dz * dz + dw * dw;
    }
#pragma unroll
    for (int off = 1; off < 32; off <<= 1) vs += __shfl_xor(vs, off);
    const float inv = 1.0f / sqrtf(vs * (1.0f / 511.0f));   // ddof=1, no eps

    const float4* alp = (const float4*)alpha;
    const float4* bep = (const float4*)beta;
    float4* op = (float4*)(out + row * 512);
#pragma unroll
    for (int q = 0; q < 4; ++q) {
        float4 al = alp[q * 32 + lane];
        float4 be = bep[q * 32 + lane];
        float4 y;
        y.x = (x[q].x - mean) * inv * al.x + be.x;
        y.y = (x[q].y - mean) * inv * al.y + be.y;
        y.z = (x[q].z - mean) * inv * al.z + be.z;
        y.w = (x[q].w - mean) * inv * al.w + be.w;
        op[q * 32 + lane] = y;
    }
}

// ---------------------------------------------------------------------------
// Host orchestration
// ---------------------------------------------------------------------------
extern "C" void kernel_launch(void* const* d_in, const int* in_sizes, int n_in,
                              void* d_out, int out_size, void* d_ws, size_t ws_size,
                              hipStream_t stream)
{
    (void)in_sizes; (void)n_in; (void)out_size; (void)ws_size;

    constexpr int L = 6, Dm = 512, FF = 2048, M = 4096;   // M = B*S
    constexpr long long DD  = (long long)Dm * Dm;         // 262144
    constexpr long long DFF = (long long)Dm * FF;         // 1048576
    const float scale = 0.04419417382415922f;             // 1/sqrt(512)

    const float* in_enc = (const float*)d_in[0];
    const float* in_dec = (const float*)d_in[1];
    const float* enc_W  = (const float*)d_in[2];
    const float* enc_b  = (const float*)d_in[3];
    const float* enc_W1 = (const float*)d_in[4];
    const float* enc_b1 = (const float*)d_in[5];
    const float* enc_W2 = (const float*)d_in[6];
    const float* enc_b2 = (const float*)d_in[7];
    const float* enc_al = (const float*)d_in[8];
    const float* enc_be = (const float*)d_in[9];
    const float* dec_W  = (const float*)d_in[10];
    const float* dec_b  = (const float*)d_in[11];
    const float* dec_W1 = (const float*)d_in[12];
    const float* dec_b1 = (const float*)d_in[13];
    const float* dec_W2 = (const float*)d_in[14];
    const float* dec_b2 = (const float*)d_in[15];
    const float* dec_al = (const float*)d_in[16];
    const float* dec_be = (const float*)d_in[17];
    const float* fin_al = (const float*)d_in[18];
    const float* fin_be = (const float*)d_in[19];
    float* outF = (float*)d_out;

    // workspace layout (floats): ~101 MB total
    float* ws   = (float*)d_ws;
    float* zBuf = ws;                        // M*D   = 2,097,152
    float* aBuf = zBuf + (size_t)M * Dm;     // M*D
    float* zenc = aBuf + (size_t)M * Dm;     // M*D
    float* z1   = zenc + (size_t)M * Dm;     // M*D
    float* z2   = z1   + (size_t)M * Dm;     // M*D
    float* qkv  = z2   + (size_t)M * Dm;     // M*3D  = 6,291,456
    float* hBuf = qkv  + (size_t)M * 3 * Dm; // M*FF  = 8,388,608 (also cross-KV)

    auto gemm = [&](const float* A, int lda, const float* W, long long pitch,
                    int chunkCols, const float* bias, float* C,
                    int N, int K, int relu) {
        dim3 grid(N / BN, M / BM);
        k_gemm<<<grid, dim3(256), 0, stream>>>(A, lda, W, pitch, chunkCols,
                                               bias, C, N, K, relu);
    };
    auto attn = [&](const float* q, int qs, const float* k, const float* v,
                    int kvs, float* out) {
        k_attn<<<dim3(32, 8, 8), dim3(32), 0, stream>>>(q, qs, k, v, kvs,
                                                        out, 512, scale);
    };
    auto addln = [&](const float* a, const float* r, const float* al,
                     const float* be, float* out, int hasR) {
        k_add_ln<<<dim3(512), dim3(256), 0, stream>>>(a, r, al, be, out, hasR);
    };

    // ===================== Encoder =====================
    const float* zCur = in_enc;
    for (int i = 0; i < L; ++i) {
        const float* Wi = enc_W + (size_t)i * 4 * DD;
        const float* bi = enc_b + (size_t)i * 4 * Dm;
        // packed QKV projection (N=1536 across three [D,D] chunks)
        gemm(zCur, Dm, Wi, DD, Dm, bi, qkv, 3 * Dm, Dm, 0);
        attn(qkv, 3 * Dm, qkv + Dm, qkv + 2 * Dm, 3 * Dm, aBuf);
        // out projection (reuse qkv buffer as dense M x D output)
        gemm(aBuf, Dm, Wi + 3 * DD, DD, Dm, bi + 3 * Dm, qkv, Dm, Dm, 0);
        addln(qkv, zCur, enc_al + (size_t)(i * 2) * Dm,
              enc_be + (size_t)(i * 2) * Dm, zBuf, 1);
        zCur = zBuf;
        // FFN
        gemm(zCur, Dm, enc_W1 + (size_t)i * DFF, DFF, FF,
             enc_b1 + (size_t)i * FF, hBuf, FF, Dm, 1);
        gemm(hBuf, FF, enc_W2 + (size_t)i * DFF, DFF, Dm,
             enc_b2 + (size_t)i * Dm, aBuf, Dm, FF, 0);
        addln(aBuf, zCur, enc_al + (size_t)(i * 2 + 1) * Dm,
              enc_be + (size_t)(i * 2 + 1) * Dm, zBuf, 1);
        zCur = zBuf;
    }
    addln(zCur, zCur, fin_al, fin_be, zenc, 0);   // z_enc

    // ===================== Decoder =====================
    zCur = in_dec;
    for (int i = 0; i < L; ++i) {
        const float* Ws = dec_W + ((size_t)i * 2 + 0) * 4 * DD;
        const float* bs = dec_b + ((size_t)i * 2 + 0) * 4 * Dm;
        const float* Wc = dec_W + ((size_t)i * 2 + 1) * 4 * DD;
        const float* bc = dec_b + ((size_t)i * 2 + 1) * 4 * Dm;

        // --- self attention ---
        gemm(zCur, Dm, Ws, DD, Dm, bs, qkv, 3 * Dm, Dm, 0);
        attn(qkv, 3 * Dm, qkv + Dm, qkv + 2 * Dm, 3 * Dm, aBuf);
        gemm(aBuf, Dm, Ws + 3 * DD, DD, Dm, bs + 3 * Dm, qkv, Dm, Dm, 0);
        addln(qkv, zCur, dec_al + (size_t)(i * 3) * Dm,
              dec_be + (size_t)(i * 3) * Dm, z1, 1);

        // --- cross attention (q from z1, kv from z_enc) ---
        gemm(z1, Dm, Wc, DD, Dm, bc, z2, Dm, Dm, 0);                   // Q -> z2
        gemm(zenc, Dm, Wc + DD, DD, Dm, bc + Dm, hBuf, 2 * Dm, Dm, 0); // KV packed
        attn(z2, Dm, hBuf, hBuf + Dm, 2 * Dm, aBuf);
        gemm(aBuf, Dm, Wc + 3 * DD, DD, Dm, bc + 3 * Dm, qkv, Dm, Dm, 0);
        addln(qkv, z1, dec_al + (size_t)(i * 3 + 1) * Dm,
              dec_be + (size_t)(i * 3 + 1) * Dm, z2, 1);

        // --- FFN ---
        gemm(z2, Dm, dec_W1 + (size_t)i * DFF, DFF, FF,
             dec_b1 + (size_t)i * FF, hBuf, FF, Dm, 1);
        gemm(hBuf, FF, dec_W2 + (size_t)i * DFF, DFF, Dm,
             dec_b2 + (size_t)i * Dm, aBuf, Dm, FF, 0);
        addln(aBuf, z2, dec_al + (size_t)(i * 3 + 2) * Dm,
              dec_be + (size_t)(i * 3 + 2) * Dm, zBuf, 1);
        zCur = zBuf;
    }
    addln(zCur, zCur, fin_al + Dm, fin_be + Dm, outF, 0);
}
